// TransformerBlock_24378234372552
// MI455X (gfx1250) — matrix operand order repaired and checked by reading, not compiled, not run on hardware
//
#include <hip/hip_runtime.h>

typedef __attribute__((ext_vector_type(16))) _Float16 v16h;
typedef __attribute__((ext_vector_type(8)))  _Float16 v8h;
typedef __attribute__((ext_vector_type(8)))  float    v8f;
typedef __attribute__((ext_vector_type(4)))  int      v4i;

typedef __attribute__((address_space(1))) v4i g_v4i;
typedef __attribute__((address_space(3))) v4i l_v4i;

#define CAT16(a,b) __builtin_shufflevector((a),(b),0,1,2,3,4,5,6,7,8,9,10,11,12,13,14,15)

__device__ __forceinline__ v16h ld_frag(const _Float16* p0, const _Float16* p1) {
    v8h a = *(const v8h*)p0;
    v8h b = *(const v8h*)p1;
    return CAT16(a, b);
}

__device__ __forceinline__ v8f wmma16(v16h a, v16h b, v8f c) {
    return __builtin_amdgcn_wmma_f32_16x16x32_f16(false, a, false, b, (short)0, c, false, false);
}

// ---------------------------------------------------------------------------
// Async global -> LDS copy of 16 bytes (8 f16). Uses gfx1250
// GLOBAL_LOAD_ASYNC_TO_LDS_B128 when the builtin exists; sync copy otherwise.
// ---------------------------------------------------------------------------
#if __has_builtin(__builtin_amdgcn_global_load_async_to_lds_b128)
#define HAVE_ASYNC_LDS 1
#else
#define HAVE_ASYNC_LDS 0
#endif

__device__ __forceinline__ void cp16_async(_Float16* lds, const _Float16* g) {
#if HAVE_ASYNC_LDS
    __builtin_amdgcn_global_load_async_to_lds_b128(
        (g_v4i*)g, (l_v4i*)lds, 0, 0);
#else
    *(uint4*)lds = *(const uint4*)g;
#endif
}

__device__ __forceinline__ void async_join() {
#if HAVE_ASYNC_LDS
#if __has_builtin(__builtin_amdgcn_s_wait_asynccnt)
    __builtin_amdgcn_s_wait_asynccnt(0);
#else
    asm volatile("s_wait_asynccnt 0x0" ::: "memory");
#endif
#endif
}

// ---------------------------------------------------------------------------
// Problem constants (fixed by the reference)
// ---------------------------------------------------------------------------
constexpr int C_B = 2, C_S = 2048, C_D = 1024, C_H = 16, C_F = 4096;
constexpr int C_BS = C_B * C_S;       // 4096 rows
constexpr int C_SLOG = 11;            // log2(S)

// ---------------------------------------------------------------------------
// Elementwise conversions
// ---------------------------------------------------------------------------
__global__ void cvt_f16_kernel(const float* __restrict__ in, _Float16* __restrict__ out, size_t n) {
    size_t i = (size_t)blockIdx.x * blockDim.x + threadIdx.x;
    if (i < n) out[i] = (_Float16)in[i];
}

// out[n][k] = in[k][n]  (weights pre-transposed so GEMM B-tiles are K-contiguous)
__global__ void cvt_t_f16_kernel(const float* __restrict__ in, _Float16* __restrict__ out,
                                 int Kd, int Nd) {
    size_t i = (size_t)blockIdx.x * blockDim.x + threadIdx.x;
    if (i >= (size_t)Kd * Nd) return;
    int k = (int)(i / Nd);
    int n = (int)(i % Nd);
    out[(size_t)n * Kd + k] = (_Float16)in[i];
}

// segment ids: cumsum over (tokens == start_id) along s, per batch
__global__ void seg_kernel(const int* __restrict__ tokens, const int* __restrict__ start_id,
                           int* __restrict__ seg, int Bn, int Sn) {
    int b = blockIdx.x * blockDim.x + threadIdx.x;
    if (b >= Bn) return;
    int st = start_id[0];
    int c = 0;
    for (int s = 0; s < Sn; ++s) {
        c += (tokens[b * Sn + s] == st) ? 1 : 0;
        seg[b * Sn + s] = c;
    }
}

// ---------------------------------------------------------------------------
// Tiled WMMA GEMM:  C[M][N] = A[M][K] * Bt[N][K]^T
// 256 threads (8 waves), tile 128x128, K-chunk 32, double-buffered LDS with
// async global->LDS tile loads.
// ---------------------------------------------------------------------------
#define BM 128
#define BN 128
#define BK 32
#define LDA 40
#define LDB 40

enum { MODE_F32 = 0, MODE_BIAS_F32 = 1, MODE_RELU_F16 = 2, MODE_KQV = 3 };

__global__ void __launch_bounds__(256)
gemm_wmma_kernel(const _Float16* __restrict__ A, const _Float16* __restrict__ Bt,
                 int M, int N, int K, int mode,
                 const float* __restrict__ bias,
                 float* __restrict__ Cf32, _Float16* __restrict__ Cf16,
                 _Float16* __restrict__ Kh, _Float16* __restrict__ Qh, _Float16* __restrict__ Vh) {
    __shared__ _Float16 Asl[2][BM * LDA];
    __shared__ _Float16 Bsl[2][BN * LDB];

    const int t    = threadIdx.x;
    const int bn   = blockIdx.x * BN;
    const int bm   = blockIdx.y * BM;
    const int wid  = t >> 5;
    const int lane = t & 31;
    const int lo   = lane & 15;
    const int hi   = lane >> 4;
    const int wm   = (wid >> 2) * 64;  // 0 or 64
    const int wn   = (wid & 3) * 32;   // 0,32,64,96

    v8f acc[4][2];
    for (int i = 0; i < 4; ++i)
        for (int j = 0; j < 2; ++j)
            acc[i][j] = (v8f)(0.0f);

    const int grow = t >> 1;          // 0..127
    const int gk   = (t & 1) * 16;    // 0 or 16
    const _Float16* ga = A + (size_t)(bm + grow) * K + gk;
    const _Float16* gb = Bt + (size_t)(bn + grow) * K + gk;
    const int lofs = grow * LDA + gk;

    auto issue_tile = [&](int kb, int st) {
        cp16_async(&Asl[st][lofs],     ga + kb);
        cp16_async(&Asl[st][lofs + 8], ga + kb + 8);
        cp16_async(&Bsl[st][lofs],     gb + kb);
        cp16_async(&Bsl[st][lofs + 8], gb + kb + 8);
    };

    const int T = K / BK;
    issue_tile(0, 0);
    async_join();
    __syncthreads();

    for (int i = 0; i < T; ++i) {
        const int st = i & 1;
        if (i + 1 < T) issue_tile((i + 1) * BK, st ^ 1);

        v16h afr[4], bfr[2];
        for (int m = 0; m < 4; ++m) {
            const _Float16* p = &Asl[st][(wm + m * 16 + lo) * LDA];
            afr[m] = ld_frag(p + hi * 8, p + 16 + hi * 8);
        }
        for (int j = 0; j < 2; ++j) {
            const _Float16* p = &Bsl[st][(wn + j * 16 + lo) * LDB];
            bfr[j] = ld_frag(p + hi * 8, p + 16 + hi * 8);
        }
        for (int m = 0; m < 4; ++m)
            for (int j = 0; j < 2; ++j)
                acc[m][j] = wmma16(afr[m], bfr[j], acc[m][j]);

        async_join();
        __syncthreads();
    }

    // epilogue (per-column work hoisted out of the row loops)
    for (int j = 0; j < 2; ++j) {
        const int col = bn + wn + j * 16 + lo;
        int hcol = 0, tt = 0, hd = 0;
        float bv = 0.0f;
        if (mode == MODE_KQV) {
            hcol = col / 192;            // 3*HD = 192
            int rem = col - hcol * 192;
            tt = rem >> 6;               // 0=k,1=q,2=v
            hd = rem & 63;
        } else if (mode != MODE_F32) {
            bv = bias[col];
        }
        for (int m = 0; m < 4; ++m) {
            for (int r = 0; r < 8; ++r) {
                const int row = bm + wm + m * 16 + r + 8 * hi;
                float v = acc[m][j][r];
                if (mode == MODE_KQV) {
                    int b = row >> C_SLOG;
                    int s = row & (C_S - 1);
                    _Float16 hv = (_Float16)v;
                    size_t bh = (size_t)b * C_H + hcol;
                    if (tt == 0)      Kh[(bh * C_S + s) * 64 + hd] = hv;
                    else if (tt == 1) Qh[(bh * C_S + s) * 64 + hd] = hv;
                    else              Vh[(bh * 64 + hd) * (size_t)C_S + s] = hv;
                } else if (mode == MODE_RELU_F16) {
                    float z = v + bv;
                    z = z > 0.0f ? z : 0.0f;
                    Cf16[(size_t)row * N + col] = (_Float16)z;
                } else {
                    if (mode == MODE_BIAS_F32) v += bv;
                    Cf32[(size_t)row * N + col] = v;
                }
            }
        }
    }
}

// ---------------------------------------------------------------------------
// Flash attention: per block = 128 q rows of one (b,h). 8 waves, 16 q rows/wave.
// Scores via WMMA (Q 16x64 * K^T), online softmax in f32, PV via WMMA.
// K/V tiles stream in via async global->LDS.
// ---------------------------------------------------------------------------
#define AQ 128
#define AK 64
#define LQK 72

__global__ void __launch_bounds__(256)
attn_kernel(const _Float16* __restrict__ Qh, const _Float16* __restrict__ Kh,
            const _Float16* __restrict__ Vh, const int* __restrict__ seg,
            _Float16* __restrict__ Oh) {
    __shared__ _Float16 Qs[AQ * LQK];
    __shared__ _Float16 Ks[AK * LQK];       // [kv][hd]
    __shared__ _Float16 Vts[AK * LQK];      // [hd][kv]
    __shared__ _Float16 Ps[8 * 16 * LQK];   // per-wave P staging
    __shared__ int qseg[AQ];
    __shared__ int kseg[AK];

    const int t    = threadIdx.x;
    const int qb   = blockIdx.x * AQ;
    const int h    = blockIdx.y;
    const int b    = blockIdx.z;
    const size_t bh = (size_t)b * C_H + h;
    const int wid  = t >> 5;
    const int lane = t & 31;
    const int lo   = lane & 15;
    const int hi   = lane >> 4;

    // load Q tile (128 x 64) + q segment ids
    {
        int r  = t >> 1;
        int c0 = (t & 1) * 32;
        const _Float16* src = Qh + (bh * C_S + qb + r) * 64 + c0;
        _Float16* dst = Qs + r * LQK + c0;
        cp16_async(dst,      src);
        cp16_async(dst + 8,  src + 8);
        cp16_async(dst + 16, src + 16);
        cp16_async(dst + 24, src + 24);
    }
    if (t < AQ) qseg[t] = seg[b * C_S + qb + t];
    async_join();
    __syncthreads();

    // Q fragments for this wave's 16 rows (2 K-chunks of 32)
    v16h qfr[2];
    {
        const _Float16* p = Qs + (wid * 16 + lo) * LQK;
        qfr[0] = ld_frag(p + hi * 8, p + 16 + hi * 8);
        qfr[1] = ld_frag(p + 32 + hi * 8, p + 48 + hi * 8);
    }

    float mrow[8], lrow[8];
    v8f oacc[4];
    for (int r = 0; r < 8; ++r) { mrow[r] = -1e30f; lrow[r] = 0.0f; }
    for (int nt = 0; nt < 4; ++nt) oacc[nt] = (v8f)(0.0f);

    const int nkv = qb / AK + 2;  // causal: kv tiles covering [0, qb+127]

    for (int jb = 0; jb < nkv; ++jb) {
        const int j0 = jb * AK;
        __syncthreads();
        {
            int r  = t >> 2;
            int c0 = (t & 3) * 16;
            const _Float16* ks = Kh + (bh * C_S + j0 + r) * 64 + c0;
            cp16_async(Ks + r * LQK + c0,     ks);
            cp16_async(Ks + r * LQK + c0 + 8, ks + 8);
            const _Float16* vs = Vh + (bh * 64 + r) * (size_t)C_S + j0 + c0;
            cp16_async(Vts + r * LQK + c0,     vs);
            cp16_async(Vts + r * LQK + c0 + 8, vs + 8);
        }
        if (t < AK) kseg[t] = seg[b * C_S + j0 + t];
        async_join();
        __syncthreads();

        // scores S = Q K^T  (16 x 64 per wave)
        v8f sacc[4];
        for (int nt = 0; nt < 4; ++nt) {
            v8f z = (v8f)(0.0f);
            for (int c = 0; c < 2; ++c) {
                const _Float16* p = Ks + (nt * 16 + lo) * LQK + c * 32;
                v16h bfr = ld_frag(p + hi * 8, p + 16 + hi * 8);
                z = wmma16(qfr[c], bfr, z);
            }
            sacc[nt] = z;
        }

        // scale + mask, per-row tile max
        float mt[8];
        for (int r = 0; r < 8; ++r) mt[r] = -1e30f;
        for (int nt = 0; nt < 4; ++nt) {
            int kv = j0 + nt * 16 + lo;
            int ks_v = kseg[nt * 16 + lo];
            for (int r = 0; r < 8; ++r) {
                int qi = qb + wid * 16 + r + 8 * hi;
                float s = sacc[nt][r] * 0.125f;  // 1/sqrt(64)
                bool ok = (kv <= qi) && (ks_v == qseg[wid * 16 + r + 8 * hi]);
                s = ok ? s : -1e30f;
                sacc[nt][r] = s;
                mt[r] = fmaxf(mt[r], s);
            }
        }
        for (int m = 1; m < 16; m <<= 1)
            for (int r = 0; r < 8; ++r)
                mt[r] = fmaxf(mt[r], __shfl_xor(mt[r], m, 32));

        float alpha[8];
        for (int r = 0; r < 8; ++r) {
            float mn = fmaxf(mrow[r], mt[r]);
            alpha[r] = __expf(mrow[r] - mn);
            mrow[r] = mn;
        }
        for (int nt = 0; nt < 4; ++nt)
            for (int r = 0; r < 8; ++r)
                oacc[nt][r] *= alpha[r];

        float psum[8] = {0, 0, 0, 0, 0, 0, 0, 0};
        _Float16* PW = Ps + wid * 16 * LQK;
        for (int nt = 0; nt < 4; ++nt) {
            for (int r = 0; r < 8; ++r) {
                float s = sacc[nt][r];
                float p = (s < -1e29f) ? 0.0f : __expf(s - mrow[r]);
                psum[r] += p;
                PW[(r + 8 * hi) * LQK + nt * 16 + lo] = (_Float16)p;
            }
        }
        for (int m = 1; m < 16; m <<= 1)
            for (int r = 0; r < 8; ++r)
                psum[r] += __shfl_xor(psum[r], m, 32);
        for (int r = 0; r < 8; ++r)
            lrow[r] = lrow[r] * alpha[r] + psum[r];

        __syncthreads();  // P visible for fragment reload

        // O += P V   (P 16x64, V 64x64)
        for (int c = 0; c < 2; ++c) {
            const _Float16* pp = PW + lo * LQK + c * 32;
            v16h afr = ld_frag(pp + hi * 8, pp + 16 + hi * 8);
            for (int nt = 0; nt < 4; ++nt) {
                const _Float16* vp = Vts + (nt * 16 + lo) * LQK + c * 32;
                v16h bfr = ld_frag(vp + hi * 8, vp + 16 + hi * 8);
                oacc[nt] = wmma16(afr, bfr, oacc[nt]);
            }
        }
    }

    // write O / l  ->  attn f16 [b][s][h*64+hd]
    for (int r = 0; r < 8; ++r) {
        float inv = lrow[r] > 0.0f ? 1.0f / lrow[r] : 0.0f;
        int qi = qb + wid * 16 + r + 8 * hi;
        for (int nt = 0; nt < 4; ++nt) {
            int d = h * 64 + nt * 16 + lo;
            Oh[((size_t)b * C_S + qi) * C_D + d] = (_Float16)(oacc[nt][r] * inv);
        }
    }
}

// ---------------------------------------------------------------------------
// LayerNorm + residual:  out = LN(in)*g + be + res ; optional f16 copy
// one block per row, 256 threads, D = 1024
// ---------------------------------------------------------------------------
__global__ void __launch_bounds__(256)
ln_res_kernel(const float* __restrict__ in, const float* __restrict__ res,
              const float* __restrict__ g, const float* __restrict__ be,
              float* __restrict__ out_f32, _Float16* __restrict__ out_f16, int Dd) {
    const int row  = blockIdx.x;
    const int t    = threadIdx.x;
    const int wid  = t >> 5;
    const int lane = t & 31;
    const float* x = in + (size_t)row * Dd;

    float v[4];
    float s = 0.0f, s2 = 0.0f;
    for (int i = 0; i < 4; ++i) {
        v[i] = x[t + 256 * i];
        s += v[i];
        s2 += v[i] * v[i];
    }
    for (int m = 1; m < 32; m <<= 1) {
        s  += __shfl_xor(s, m, 32);
        s2 += __shfl_xor(s2, m, 32);
    }
    __shared__ float ws1[8], ws2[8];
    if (lane == 0) { ws1[wid] = s; ws2[wid] = s2; }
    __syncthreads();
    if (t == 0) {
        float a = 0.0f, b2 = 0.0f;
        for (int i = 0; i < 8; ++i) { a += ws1[i]; b2 += ws2[i]; }
        ws1[0] = a; ws2[0] = b2;
    }
    __syncthreads();
    float mean = ws1[0] / Dd;
    float var  = ws2[0] / Dd - mean * mean;
    float rstd = rsqrtf(var + 1e-6f);
    for (int i = 0; i < 4; ++i) {
        int c = t + 256 * i;
        float y = (v[i] - mean) * rstd * g[c] + be[c] + res[(size_t)row * Dd + c];
        out_f32[(size_t)row * Dd + c] = y;
        if (out_f16) out_f16[(size_t)row * Dd + c] = (_Float16)y;
    }
}

// ---------------------------------------------------------------------------
// host launcher
// ---------------------------------------------------------------------------
extern "C" void kernel_launch(void* const* d_in, const int* in_sizes, int n_in,
                              void* d_out, int out_size, void* d_ws, size_t ws_size,
                              hipStream_t stream) {
    (void)in_sizes; (void)n_in; (void)out_size; (void)ws_size;

    const float* x_embeds = (const float*)d_in[0];
    const int*   tokens   = (const int*)d_in[1];
    const float* KQV      = (const float*)d_in[2];
    const float* WO       = (const float*)d_in[3];
    const float* W_up     = (const float*)d_in[4];
    const float* b_up     = (const float*)d_in[5];
    const float* W_down   = (const float*)d_in[6];
    const float* b_down   = (const float*)d_in[7];
    const float* ln1_g    = (const float*)d_in[8];
    const float* ln1_b    = (const float*)d_in[9];
    const float* ln2_g    = (const float*)d_in[10];
    const float* ln2_b    = (const float*)d_in[11];
    const int*   start_id = (const int*)d_in[12];
    float* out = (float*)d_out;

    constexpr int Bc = C_B, Sc = C_S, Dc = C_D, Hc = C_H, Fc = C_F;
    constexpr int BS = C_BS;

    char* w = (char*)d_ws;
    auto alloc = [&](size_t bytes) -> char* {
        char* p = w;
        w += (bytes + 255) & ~(size_t)255;
        return p;
    };
    _Float16* Xh    = (_Float16*)alloc((size_t)BS * Dc * 2);
    _Float16* KQVt  = (_Float16*)alloc((size_t)3 * Dc * Dc * 2);
    _Float16* WOt   = (_Float16*)alloc((size_t)Dc * Dc * 2);
    _Float16* Wupt  = (_Float16*)alloc((size_t)Fc * Dc * 2);
    _Float16* Wdnt  = (_Float16*)alloc((size_t)Dc * Fc * 2);
    _Float16* Qh    = (_Float16*)alloc((size_t)BS * Dc * 2);
    _Float16* Kh    = (_Float16*)alloc((size_t)BS * Dc * 2);
    _Float16* Vh    = (_Float16*)alloc((size_t)BS * Dc * 2);
    _Float16* attnh = (_Float16*)alloc((size_t)BS * Dc * 2);
    float*    proj  = (float*)alloc((size_t)BS * Dc * 4);
    float*    xres  = (float*)alloc((size_t)BS * Dc * 4);
    _Float16* xh    = (_Float16*)alloc((size_t)BS * Dc * 2);
    _Float16* hh    = (_Float16*)alloc((size_t)BS * Fc * 2);
    float*    mlp   = (float*)alloc((size_t)BS * Dc * 4);
    int*      segb  = (int*)alloc((size_t)BS * 4);

    // 1) conversions
    {
        size_t n = (size_t)BS * Dc;
        cvt_f16_kernel<<<(unsigned)((n + 255) / 256), 256, 0, stream>>>(x_embeds, Xh, n);
    }
    cvt_t_f16_kernel<<<(Dc * 3 * Dc + 255) / 256, 256, 0, stream>>>(KQV, KQVt, Dc, 3 * Dc);
    cvt_t_f16_kernel<<<(Dc * Dc + 255) / 256, 256, 0, stream>>>(WO, WOt, Dc, Dc);
    cvt_t_f16_kernel<<<(Dc * Fc + 255) / 256, 256, 0, stream>>>(W_up, Wupt, Dc, Fc);
    cvt_t_f16_kernel<<<(Fc * Dc + 255) / 256, 256, 0, stream>>>(W_down, Wdnt, Fc, Dc);
    seg_kernel<<<1, 32, 0, stream>>>(tokens, start_id, segb, Bc, Sc);

    // 2) KQV projection, scattered into head layouts
    gemm_wmma_kernel<<<dim3(3 * Dc / BN, BS / BM), 256, 0, stream>>>(
        Xh, KQVt, BS, 3 * Dc, Dc, MODE_KQV, nullptr,
        nullptr, nullptr, Kh, Qh, Vh);

    // 3) attention
    attn_kernel<<<dim3(Sc / AQ, Hc, Bc), 256, 0, stream>>>(Qh, Kh, Vh, segb, attnh);

    // 4) output projection (no bias) -> proj f32
    gemm_wmma_kernel<<<dim3(Dc / BN, BS / BM), 256, 0, stream>>>(
        attnh, WOt, BS, Dc, Dc, MODE_F32, nullptr,
        proj, nullptr, nullptr, nullptr, nullptr);

    // 5) LN1 + residual: x = LN(proj) + x_embeds  (also f16 copy)
    ln_res_kernel<<<BS, 256, 0, stream>>>(proj, x_embeds, ln1_g, ln1_b, xres, xh, Dc);

    // 6) MLP up: relu(x @ W_up + b_up) -> f16
    gemm_wmma_kernel<<<dim3(Fc / BN, BS / BM), 256, 0, stream>>>(
        xh, Wupt, BS, Fc, Dc, MODE_RELU_F16, b_up,
        nullptr, hh, nullptr, nullptr, nullptr);

    // 7) MLP down: h @ W_down + b_down -> f32
    gemm_wmma_kernel<<<dim3(Dc / BN, BS / BM), 256, 0, stream>>>(
        hh, Wdnt, BS, Dc, Fc, MODE_BIAS_F32, b_down,
        mlp, nullptr, nullptr, nullptr, nullptr);

    // 8) LN2 + residual: out = LN(mlp) + x
    ln_res_kernel<<<BS, 256, 0, stream>>>(mlp, xres, ln2_g, ln2_b, out, nullptr, Dc);
}
